// Bert_BiLSTM_CRF_81716047774105
// MI455X (gfx1250) — matrix operand order repaired and checked by reading, not compiled, not run on hardware
//
#include <hip/hip_runtime.h>
#include <hip/hip_bf16.h>
#include <math.h>

// ---------------------------------------------------------------------------
// Bert-BiLSTM-CRF for MI455X (gfx1250, wave32, WMMA + TDM).
// Heavy GEMMs run on v_wmma_f32_16x16x32_bf16 (bf16 in, fp32 accumulate).
// W_hh slices are staged into LDS by the Tensor Data Mover.
// ---------------------------------------------------------------------------

typedef __attribute__((ext_vector_type(16))) __bf16 v16bf;
typedef __attribute__((ext_vector_type(8)))  float  v8f;
typedef __attribute__((ext_vector_type(4)))  unsigned int v4u;
typedef __attribute__((ext_vector_type(8)))  int v8i;
typedef __attribute__((ext_vector_type(4)))  int v4i;

#define H_BERT 768
#define H_LSTM 384
#define G4     1536          // 4*H_LSTM (gates per direction)
#define NGC    3072          // both directions concatenated
#define NTAGS  9
#define BB     64
#define SS     512
#define NROWS  (BB*SS)       // 32768

// dynamic-LDS layout for k_lstm (bytes)
#define LDS_WSH   0                     // 96*384 bf16 = 73728 B
#define LDS_GBUF  73728                 // 64*96 fp32  = 24576 B
#define LDS_CST   (73728 + 24576)       // 64*24 fp32  =  6144 B
#define LDS_TOTAL (73728 + 24576 + 6144)

union Frag16 { v16bf v; uint4 q[2]; };

__device__ __forceinline__ unsigned short f2bf(float x) {
  unsigned u = __float_as_uint(x);
  unsigned r = (u + 0x7FFFu + ((u >> 16) & 1u)) >> 16;   // RNE
  return (unsigned short)r;
}
__device__ __forceinline__ float bf2f(unsigned short h) {
  return __uint_as_float(((unsigned)h) << 16);
}
__device__ __forceinline__ float sigmoidf_(float x) {
  return 1.f / (1.f + __expf(-x));
}

// ---------------------------------------------------------------------------
// Prep: weights fp32 -> bf16, fused bias, zero h ping-pong + barrier counters.
// ---------------------------------------------------------------------------
__global__ void __launch_bounds__(256) k_prep(
    const float* __restrict__ wihf, const float* __restrict__ wihb,
    const float* __restrict__ whhf, const float* __restrict__ whhb,
    const float* __restrict__ bihf, const float* __restrict__ bhhf,
    const float* __restrict__ bihb, const float* __restrict__ bhhb,
    unsigned short* __restrict__ wih, unsigned short* __restrict__ whh,
    float* __restrict__ bias, unsigned short* __restrict__ hping,
    unsigned* __restrict__ bar)
{
  int i = blockIdx.x * 256 + threadIdx.x;
  if (i < NGC * H_BERT) {
    int r = i / H_BERT, c = i % H_BERT;
    float v = (r < G4) ? wihf[r * H_BERT + c] : wihb[(r - G4) * H_BERT + c];
    wih[i] = f2bf(v);
  }
  if (i < NGC * H_LSTM) {
    int r = i / H_LSTM, c = i % H_LSTM;
    float v = (r < G4) ? whhf[r * H_LSTM + c] : whhb[(r - G4) * H_LSTM + c];
    whh[i] = f2bf(v);
  }
  if (i < NGC) bias[i] = (i < G4) ? (bihf[i] + bhhf[i]) : (bihb[i - G4] + bhhb[i - G4]);
  if (i < 4 * BB * H_LSTM) hping[i] = 0;     // 2 dirs x 2 ping-pong buffers
  if (i < 4) bar[i] = 0;                     // device-barrier counters
}

// embeds fp32 -> bf16, 8 elements per thread (grid*256*8 == 25,165,824 exactly)
__global__ void __launch_bounds__(256) k_embconv(
    const float* __restrict__ x, unsigned short* __restrict__ xb)
{
  size_t i = ((size_t)blockIdx.x * 256 + threadIdx.x) * 8;
  float4 a = *(const float4*)(x + i);
  float4 b = *(const float4*)(x + i + 4);
  uint4 pk;
  pk.x = (unsigned)f2bf(a.x) | ((unsigned)f2bf(a.y) << 16);
  pk.y = (unsigned)f2bf(a.z) | ((unsigned)f2bf(a.w) << 16);
  pk.z = (unsigned)f2bf(b.x) | ((unsigned)f2bf(b.y) << 16);
  pk.w = (unsigned)f2bf(b.z) | ((unsigned)f2bf(b.w) << 16);
  *(uint4*)(xb + i) = pk;
}

// ---------------------------------------------------------------------------
// xs = embeds @ W_ih^T + (b_ih + b_hh), both directions at once.
// M=32768, K=768, N=3072. One wave computes a 64x32 output block
// (4x2 tiles of 16x16) -> 8 wmma per K=32 step, 24 K-steps.
// Output stored transposed: xs[s][b][3072] (recurrence-friendly).
// ---------------------------------------------------------------------------
__global__ void __launch_bounds__(256) k_gemm_xs(
    const unsigned short* __restrict__ A,   // [32768][768] bf16 (row = b*512+s)
    const unsigned short* __restrict__ Bm,  // [3072][768]  bf16 (row = out col)
    const float* __restrict__ bias,         // [3072]
    float* __restrict__ xs)                 // [512][64][3072] fp32
{
  const int lane = threadIdx.x & 31;
  const int w    = threadIdx.x >> 5;
  const int tile = blockIdx.x * 8 + w;      // 49152 tiles total
  const int tm = tile / 96, tn = tile % 96;
  const int row0 = tm * 64, col0 = tn * 32;
  const int ln = lane & 15, sect = lane >> 4;

  v8f acc[4][2];
  #pragma unroll
  for (int mi = 0; mi < 4; ++mi)
    #pragma unroll
    for (int ni = 0; ni < 2; ++ni)
      #pragma unroll
      for (int r = 0; r < 8; ++r) acc[mi][ni][r] = 0.f;

  const unsigned short* arow[4];
  #pragma unroll
  for (int mi = 0; mi < 4; ++mi)
    arow[mi] = A + (size_t)(row0 + mi * 16 + ln) * H_BERT;
  const unsigned short* brow[2];
  #pragma unroll
  for (int ni = 0; ni < 2; ++ni)
    brow[ni] = Bm + (size_t)(col0 + ni * 16 + ln) * H_BERT;

  for (int kk = 0; kk < H_BERT; kk += 32) {
    Frag16 a[4], b[2];
    const int c0 = kk + sect * 8;           // A: lanes hold K {c0..c0+7, c0+16..c0+23}
    #pragma unroll
    for (int mi = 0; mi < 4; ++mi) {
      a[mi].q[0] = *(const uint4*)(arow[mi] + c0);
      a[mi].q[1] = *(const uint4*)(arow[mi] + c0 + 16);
    }
    const int c1 = kk + sect * 8;           // B: lanes hold K {c1..c1+7, c1+16..c1+23}
    #pragma unroll
    for (int ni = 0; ni < 2; ++ni) {
      b[ni].q[0] = *(const uint4*)(brow[ni] + c1);
      b[ni].q[1] = *(const uint4*)(brow[ni] + c1 + 16);
    }
    if (kk + 64 < H_BERT) {
      __builtin_prefetch(arow[0] + kk + 64, 0, 1);
      __builtin_prefetch(brow[0] + kk + 64, 0, 1);
    }
    #pragma unroll
    for (int mi = 0; mi < 4; ++mi)
      #pragma unroll
      for (int ni = 0; ni < 2; ++ni)
        acc[mi][ni] = __builtin_amdgcn_wmma_f32_16x16x32_bf16(
            false, a[mi].v, false, b[ni].v, (short)0, acc[mi][ni], false, false);
  }

  #pragma unroll
  for (int ni = 0; ni < 2; ++ni) {
    const int col = col0 + ni * 16 + ln;
    const float bv = bias[col];
    #pragma unroll
    for (int mi = 0; mi < 4; ++mi) {
      #pragma unroll
      for (int r = 0; r < 8; ++r) {
        const int row = row0 + mi * 16 + sect * 8 + r;  // = b*512 + s
        const int b_ = row >> 9, s_ = row & 511;
        xs[((size_t)s_ * BB + b_) * NGC + col] = acc[mi][ni][r] + bv;
      }
    }
  }
}

// ---------------------------------------------------------------------------
// Persistent BiLSTM recurrence. 32 workgroups: 16 per direction, each owns
// 24 hidden units (4 gate rows each -> 96 W_hh rows resident in LDS, bf16,
// staged by the Tensor Data Mover: 4 x 2D tiles of 24 rows x 384 cols).
// Per step: WMMA GEMM M=64 N=96 K=384, gates, h -> bf16 ping-pong buffer,
// device-scope barrier (atomic + s_sleep spin) across the 16 WGs of a dir.
// ---------------------------------------------------------------------------
__global__ void __launch_bounds__(256) k_lstm(
    const float* __restrict__ xs,            // [512][64][3072]
    const unsigned short* __restrict__ whh,  // [3072][384] bf16
    unsigned short* __restrict__ hping,      // [2 dirs][2 bufs][64][384] bf16
    unsigned short* __restrict__ hout,       // [2 dirs][512][64][384] bf16
    unsigned* __restrict__ bar)
{
  extern __shared__ char smem[];
  unsigned short* Wsh = (unsigned short*)(smem + LDS_WSH);
  float* gbuf = (float*)(smem + LDS_GBUF);
  float* cst  = (float*)(smem + LDS_CST);

  const int tid  = threadIdx.x;
  const int dir  = blockIdx.x >> 4;
  const int wg   = blockIdx.x & 15;
  const int lane = tid & 31, w = tid >> 5;
  const int ln = lane & 15, sect = lane >> 4;

  // --- TDM: stage W_hh slice into LDS. Wave 0 only (scalar branch via
  // readfirstlane; TDM ignores EXEC so lane-masking would not prevent issue).
  if (__builtin_amdgcn_readfirstlane(tid >> 5) == 0) {
    #pragma unroll
    for (int gate = 0; gate < 4; ++gate) {
      const int grow0 = dir * G4 + gate * H_LSTM + wg * 24;     // first W_hh row
      const unsigned long long ga =
          (unsigned long long)(const void*)whh + (unsigned long long)grow0 * (H_LSTM * 2);
      // D# group 0: count=1 | lds_addr | global_addr(57b) | type=2
      v4u g0;
      g0.x = 1u;                                        // count=1, user mode
      g0.y = (unsigned)(LDS_WSH + gate * 24 * H_LSTM * 2);
      g0.z = (unsigned)(ga & 0xFFFFFFFFull);
      g0.w = (unsigned)((ga >> 32) & 0x01FFFFFFull) | (2u << 30);  // type=2
      // D# group 1: data_size=2B; tensor 384 x 24 rows, stride 384;
      // tile_dim0=384 (row), tile_dim1=24 (rows)
      v8i g1;
      g1[0] = 0x10000;                 // data_size=1 (2 bytes/elem)
      g1[1] = (int)(384u << 16);       // tensor_dim0[15:0] in bits[31:16]
      g1[2] = (int)(24u << 16);        // tensor_dim0 hi=0 | tensor_dim1=24
      g1[3] = (int)(384u << 16);       // tensor_dim1 hi=0 | tile_dim0=384
      g1[4] = 24;                      // tile_dim1=24, tile_dim2=0
      g1[5] = 384;                     // tensor_dim0_stride lo
      g1[6] = 0;                       // stride hi / dim1_stride lo
      g1[7] = 0;
      v4i z4 = {0, 0, 0, 0};           // groups 2/3 unused (2D tensor)
      v8i z8 = {0, 0, 0, 0, 0, 0, 0, 0};
      __builtin_amdgcn_tensor_load_to_lds(g0, g1, z4, z4, z8, 0);
    }
    __builtin_amdgcn_s_wait_tensorcnt(0);
  }
  for (int idx = tid; idx < 64 * 24; idx += 256) cst[idx] = 0.f;
  __syncthreads();

  for (int p = 0; p < SS; ++p) {
    const int s = dir ? (SS - 1 - p) : p;
    const unsigned short* hr = hping + (size_t)(dir * 2 + (p & 1)) * BB * H_LSTM;
    unsigned short*       hw = hping + (size_t)(dir * 2 + ((p + 1) & 1)) * BB * H_LSTM;
    const float* xsrow = xs + (size_t)s * BB * NGC;

    // GEMM: 24 tiles (tm 0..3, tn 0..5), 3 per wave
    #pragma unroll
    for (int i3 = 0; i3 < 3; ++i3) {
      const int tile = w + 8 * i3;
      const int tm = tile & 3, tn = tile >> 2;
      const int ncol = tn * 16 + ln;                  // 0..95
      const int gate = ncol / 24, jj = ncol % 24;
      const int xscol = dir * G4 + gate * H_LSTM + wg * 24 + jj;

      v8f acc;
      #pragma unroll
      for (int r = 0; r < 8; ++r) {
        const int b = tm * 16 + sect * 8 + r;
        acc[r] = xsrow[(size_t)b * NGC + xscol];      // init with xs (incl. bias)
      }
      for (int kk = 0; kk < H_LSTM; kk += 32) {
        Frag16 af, bf_;
        const int bA = tm * 16 + ln;
        const unsigned short* pa = hr + (size_t)bA * H_LSTM + kk + sect * 8;
        af.q[0] = *(const uint4*)(pa);
        af.q[1] = *(const uint4*)(pa + 16);
        const unsigned short* pb = Wsh + (size_t)ncol * H_LSTM + kk + sect * 8;
        bf_.q[0] = *(const uint4*)(pb);
        bf_.q[1] = *(const uint4*)(pb + 16);
        acc = __builtin_amdgcn_wmma_f32_16x16x32_bf16(
            false, af.v, false, bf_.v, (short)0, acc, false, false);
      }
      #pragma unroll
      for (int r = 0; r < 8; ++r) {
        const int b = tm * 16 + sect * 8 + r;
        gbuf[b * 96 + ncol] = acc[r];
      }
    }
    __syncthreads();

    // gates + state update (i,f,g,o at ncol offsets 0/24/48/72)
    for (int e = tid; e < 64 * 24; e += 256) {
      const int b = e / 24, jj = e % 24;
      const float* gb = gbuf + b * 96 + jj;
      const float i_ = sigmoidf_(gb[0]);
      const float f_ = sigmoidf_(gb[24]);
      const float g_ = tanhf(gb[48]);
      const float o_ = sigmoidf_(gb[72]);
      const float c  = f_ * cst[e] + i_ * g_;
      cst[e] = c;
      const unsigned short hb = f2bf(o_ * tanhf(c));
      const int col = wg * 24 + jj;
      hw[(size_t)b * H_LSTM + col] = hb;
      hout[((size_t)dir * SS + s) * BB * H_LSTM + (size_t)b * H_LSTM + col] = hb;
    }
    __syncthreads();

    // device-scope barrier over this direction's 16 workgroups
    if (tid == 0) {
      __threadfence();
      atomicAdd(&bar[dir], 1u);
      const unsigned tgt = 16u * (unsigned)(p + 1);
      while (atomicAdd(&bar[dir], 0u) < tgt) { __builtin_amdgcn_s_sleep(1); }
    }
    __syncthreads();
  }
}

// ---------------------------------------------------------------------------
// emissions[b,s,t] = h_f . fc_w[t,:384] + h_b . fc_w[t,384:] + fc_b[t]
// ---------------------------------------------------------------------------
__global__ void __launch_bounds__(256) k_fc(
    const unsigned short* __restrict__ hout, const float* __restrict__ fcw,
    const float* __restrict__ fcb, float* __restrict__ em)
{
  const int i = blockIdx.x * 256 + threadIdx.x;
  if (i >= BB * SS * NTAGS) return;
  const int tag = i % NTAGS;
  const int bs = i / NTAGS;
  const int b = bs / SS, s = bs % SS;
  const unsigned short* hf = hout + (size_t)s * BB * H_LSTM + (size_t)b * H_LSTM;
  const unsigned short* hb = hout + ((size_t)SS + s) * BB * H_LSTM + (size_t)b * H_LSTM;
  const float* wrow = fcw + tag * H_BERT;
  float acc = fcb[tag];
  #pragma unroll 4
  for (int k = 0; k < H_LSTM; ++k) acc += bf2f(hf[k]) * wrow[k];
  #pragma unroll 4
  for (int k = 0; k < H_LSTM; ++k) acc += bf2f(hb[k]) * wrow[H_LSTM + k];
  em[i] = acc;  // i == (b*512+s)*9 + tag
}

// ---------------------------------------------------------------------------
// CRF NLL: one block, 576 threads = 64 batches x 9 tags.
// ---------------------------------------------------------------------------
__global__ void __launch_bounds__(576) k_crf(
    const float* __restrict__ em, const int* __restrict__ tags,
    const unsigned char* __restrict__ mask, const float* __restrict__ startv,
    const float* __restrict__ endv, const float* __restrict__ trans,
    float* __restrict__ out)
{
  __shared__ float sc[64][9];
  __shared__ float tr[81];
  __shared__ float st[9], en[9];
  __shared__ float part[64];

  const int tid = threadIdx.x;
  if (tid < 81) tr[tid] = trans[tid];
  if (tid < 9) { st[tid] = startv[tid]; en[tid] = endv[tid]; }
  const int b = tid / 9, j = tid % 9;
  __syncthreads();

  sc[b][j] = st[j] + em[(size_t)b * SS * 9 + j];
  __syncthreads();

  for (int t = 1; t < SS; ++t) {
    float xs_[9]; float m = -3.4e38f;
    #pragma unroll
    for (int i = 0; i < 9; ++i) { xs_[i] = sc[b][i] + tr[i * 9 + j]; m = fmaxf(m, xs_[i]); }
    float ssum = 0.f;
    #pragma unroll
    for (int i = 0; i < 9; ++i) ssum += __expf(xs_[i] - m);
    const float nxt = m + __logf(ssum) + em[((size_t)b * SS + t) * 9 + j];
    const int mk = mask[b * SS + t];
    __syncthreads();
    if (mk) sc[b][j] = nxt;
    __syncthreads();
  }

  if (tid < 64) {
    const int bb = tid;
    const int* tg = tags + bb * SS;
    const unsigned char* mk = mask + bb * SS;
    float nm = st[tg[0]] + em[(size_t)bb * SS * 9 + tg[0]];
    int cnt = mk[0] ? 1 : 0;
    int prev = tg[0];
    for (int t = 1; t < SS; ++t) {
      const int cur = tg[t];
      if (mk[t]) {
        nm += tr[prev * 9 + cur] + em[((size_t)bb * SS + t) * 9 + cur];
        ++cnt;
      }
      prev = cur;
    }
    const int lastidx = cnt - 1;
    nm += en[tg[lastidx]];
    float m = -3.4e38f;
    for (int jj = 0; jj < 9; ++jj) m = fmaxf(m, sc[bb][jj] + en[jj]);
    float ssum = 0.f;
    for (int jj = 0; jj < 9; ++jj) ssum += __expf(sc[bb][jj] + en[jj] - m);
    part[bb] = nm - (m + __logf(ssum));
  }
  __syncthreads();
  if (tid == 0) {
    float tot = 0.f;
    for (int bb = 0; bb < 64; ++bb) tot += part[bb];
    out[0] = -tot;
  }
}

// ---------------------------------------------------------------------------
extern "C" void kernel_launch(void* const* d_in, const int* in_sizes, int n_in,
                              void* d_out, int out_size, void* d_ws, size_t ws_size,
                              hipStream_t stream)
{
  (void)in_sizes; (void)n_in; (void)out_size; (void)ws_size;
  const float* embeds = (const float*)d_in[0];
  const int*   tags   = (const int*)d_in[1];
  const unsigned char* mask = (const unsigned char*)d_in[2];
  const float* wihf = (const float*)d_in[3];
  const float* whhf = (const float*)d_in[4];
  const float* bihf = (const float*)d_in[5];
  const float* bhhf = (const float*)d_in[6];
  const float* wihb = (const float*)d_in[7];
  const float* whhb = (const float*)d_in[8];
  const float* bihb = (const float*)d_in[9];
  const float* bhhb = (const float*)d_in[10];
  const float* fcw  = (const float*)d_in[11];
  const float* fcb  = (const float*)d_in[12];
  const float* startv = (const float*)d_in[13];
  const float* endv   = (const float*)d_in[14];
  const float* transv = (const float*)d_in[15];

  char* ws = (char*)d_ws;
  size_t off = 0;
  auto alloc = [&](size_t sz) { size_t o = off; off = (off + sz + 255) & ~(size_t)255; return o; };

  unsigned short* embbf = (unsigned short*)(ws + alloc((size_t)NROWS * H_BERT * 2)); // 50.3 MB
  unsigned short* wihbf = (unsigned short*)(ws + alloc((size_t)NGC * H_BERT * 2));   // 4.7 MB
  unsigned short* whhbf = (unsigned short*)(ws + alloc((size_t)NGC * H_LSTM * 2));   // 2.4 MB
  float*          bias  = (float*)(ws + alloc((size_t)NGC * 4));
  unsigned short* hping = (unsigned short*)(ws + alloc((size_t)4 * BB * H_LSTM * 2));
  unsigned short* hout  = (unsigned short*)(ws + alloc((size_t)2 * SS * BB * H_LSTM * 2)); // 50.3 MB
  float*          em    = (float*)(ws + alloc((size_t)BB * SS * NTAGS * 4));         // 1.2 MB
  unsigned*       bar   = (unsigned*)(ws + alloc(64));
  float*          xs    = (float*)(ws + alloc((size_t)SS * BB * NGC * 4));           // 402 MB

  k_prep<<<9216, 256, 0, stream>>>(wihf, wihb, whhf, whhb, bihf, bhhf, bihb, bhhb,
                                   wihbf, whhbf, bias, hping, bar);
  k_embconv<<<12288, 256, 0, stream>>>(embeds, embbf);
  k_gemm_xs<<<6144, 256, 0, stream>>>(embbf, wihbf, bias, xs);
  k_lstm<<<32, 256, LDS_TOTAL, stream>>>(xs, whhbf, hping, hout, bar);
  k_fc<<<(BB * SS * NTAGS + 255) / 256, 256, 0, stream>>>(hout, fcw, fcb, em);
  k_crf<<<1, 576, 0, stream>>>(em, tags, mask, startv, endv, transv, (float*)d_out);
}
